// GNNActor_29661044146778
// MI455X (gfx1250) — matrix operand order repaired and checked by reading, not compiled, not run on hardware
//
#include <hip/hip_runtime.h>
#include <hip/hip_bf16.h>
#include <math.h>

// ---------------------------------------------------------------------------
// GNN actor forward for MI455X (gfx1250, wave32, WMMA).
// B=256, N=512, K=16, D_IN=32, D_H=128, D_OUT=8  -> 131072 nodes.
// GEMMs via v_wmma_f32_16x16x32_f16 with transposed (NxK) f16 weights so both
// A and B fragments are contiguous 16B vector loads. 4 M-tiles per wave for
// B-fragment reuse; global_prefetch for the streaming A operand.
// kNN: one wave per node, LDS-staged positions, shfl_xor argmin merge.
// Aggregation: atomic f32 scatter (L2-resident, 192MB L2).
// ---------------------------------------------------------------------------

typedef __attribute__((ext_vector_type(16))) _Float16 v16h;
typedef __attribute__((ext_vector_type(8)))  _Float16 v8h;
typedef __attribute__((ext_vector_type(8)))  float    v8f;

#define NB   256
#define NN   512
#define KNN  16
#define DIN  32
#define DH   128
#define DOUT 8
#define NNODES (NB * NN)   // 131072

// ------------------------------- utility kernels ---------------------------

__global__ void zero_f32_kernel(float* __restrict__ p, int n) {
  int t = blockIdx.x * 256 + threadIdx.x;
  if (t < n) p[t] = 0.0f;
}

__global__ void cast_f32_f16_kernel(const float* __restrict__ src,
                                    _Float16* __restrict__ dst, int n) {
  int t = blockIdx.x * 256 + threadIdx.x;
  if (t < n) dst[t] = (_Float16)src[t];
}

// dst(N x K, f16) = transpose(src(K x N, f32))
__global__ void transpose_cast_kernel(const float* __restrict__ src,
                                      _Float16* __restrict__ dst,
                                      int K, int N) {
  int t = blockIdx.x * 256 + threadIdx.x;   // over K*N
  if (t < K * N) {
    int k = t / N, n = t % N;
    dst[(size_t)n * K + k] = (_Float16)src[t];
  }
}

// Wout (128x8 f32) -> transposed padded 16x128 f16 (rows=cols of Wout)
__global__ void pad_wout_t_kernel(const float* __restrict__ W,
                                  _Float16* __restrict__ WpT) {
  int t = blockIdx.x * 256 + threadIdx.x;   // 0..2047
  int c = t >> 7, k = t & 127;
  WpT[t] = (c < DOUT) ? (_Float16)W[k * DOUT + c] : (_Float16)0.0f;
}

__global__ void dinv_kernel(const float* __restrict__ deg,
                            float* __restrict__ dinv, int n) {
  int t = blockIdx.x * 256 + threadIdx.x;
  if (t < n) dinv[t] = rsqrtf(deg[t] + 1.0f);
}

// ------------------------------- kNN ---------------------------------------
// grid = 16384 blocks (64 per batch), block = 256 threads = 8 waves.
__global__ void knn_kernel(const float* __restrict__ obs,
                           int* __restrict__ nbr,
                           float* __restrict__ deg) {
  __shared__ float spx[NN];
  __shared__ float spy[NN];
  int batch = blockIdx.x >> 6;
  int group = blockIdx.x & 63;
  int tid   = threadIdx.x;
  int base  = batch * NN;

  for (int c = tid; c < NN; c += 256) {
    spx[c] = obs[(size_t)(base + c) * DIN + 0];
    spy[c] = obs[(size_t)(base + c) * DIN + 1];
  }
  __syncthreads();

  int wave = tid >> 5;
  int lane = tid & 31;
  int node = group * 8 + wave;
  float px = spx[node], py = spy[node];

  float dloc[16];
  #pragma unroll
  for (int t = 0; t < 16; ++t) {
    int c = lane + 32 * t;
    float dx = spx[c] - px, dy = spy[c] - py;
    float d2 = dx * dx + dy * dy;
    if (c == node) d2 = 3.4e38f;          // exclude self
    dloc[t] = d2;
  }

  unsigned used = 0;
  int mywin = 0;
  for (int r = 0; r < 16; ++r) {
    float bd = 3.39e38f; int bt = -1;
    #pragma unroll
    for (int t = 0; t < 16; ++t) {
      bool ok = ((used >> t) & 1u) == 0u;
      if (ok && dloc[t] < bd) { bd = dloc[t]; bt = t; }
    }
    int bidx = (bt >= 0) ? (lane + 32 * bt) : (1 << 30);
    if (bt < 0) bd = 3.4e38f;

    float vd = bd; int vi = bidx;
    #pragma unroll
    for (int off = 16; off > 0; off >>= 1) {
      float od = __shfl_xor(vd, off, 32);
      int   oi = __shfl_xor(vi, off, 32);
      if (od < vd || (od == vd && oi < vi)) { vd = od; vi = oi; }
    }
    if (lane == r) mywin = vi;
    if (bidx == vi && bt >= 0) used |= (1u << bt);
  }

  int gnode = base + node;
  if (lane < KNN) {
    int nb = base + mywin;
    nbr[(size_t)gnode * KNN + lane] = nb;
    __hip_atomic_fetch_add(&deg[nb], 1.0f, __ATOMIC_RELAXED,
                           __HIP_MEMORY_SCOPE_AGENT);
  }
}

// ------------------------------- WMMA fragments ----------------------------
// A: 16x32 f16 tile from row-major (stride K). Two 16B loads per lane.
__device__ __forceinline__ v16h load_a_frag(const _Float16* __restrict__ A,
                                            int K, int m0, int kb, int lane) {
  int r = lane & 15, hi = lane >> 4;
  const _Float16* p = A + (size_t)(m0 + r) * K + kb + hi * 8;
  v8h lo = *(const v8h*)(p);
  v8h up = *(const v8h*)(p + 16);
  v16h a;
  #pragma unroll
  for (int i = 0; i < 8; ++i) { a[i] = lo[i]; a[8 + i] = up[i]; }
  return a;
}

// B: 32x16 f16 tile from TRANSPOSED weights BT (N x K, stride K).
// lane c (<16): b[i] = BT[c][kb+i], b[8+i] = BT[c][kb+16+i]
// lane c+16   : b[i] = BT[c][kb+8+i], b[8+i] = BT[c][kb+24+i]
__device__ __forceinline__ v16h load_bT_frag(const _Float16* __restrict__ BT,
                                             int K, int n0, int kb, int lane) {
  int c = lane & 15, hi = lane >> 4;
  const _Float16* p = BT + (size_t)(n0 + c) * K + kb + hi * 8;
  v8h lo = *(const v8h*)(p);
  v8h up = *(const v8h*)(p + 16);
  v16h b;
  #pragma unroll
  for (int i = 0; i < 8; ++i) { b[i] = lo[i]; b[8 + i] = up[i]; }
  return b;
}

// ------------------------------- GEMM kernels ------------------------------
// C(MxN,f32) = A(MxK,f16) @ B(KxN,f16 stored transposed as BT NxK).
// One wave: MT consecutive 16-row tiles x one 16-col tile; B frags hoisted.
template <int KT, int MT>
__global__ void gemm_h_kernel(const _Float16* __restrict__ A,
                              const _Float16* __restrict__ BT,
                              float* __restrict__ C,
                              int N, int ntile_bits) {
  constexpr int K = KT * 32;
  int wave = threadIdx.x >> 5, lane = threadIdx.x & 31;
  int tile = blockIdx.x * 8 + wave;
  int mg = tile >> ntile_bits;
  int nt = tile & ((1 << ntile_bits) - 1);
  int n0 = nt * 16;
  int m0 = mg * (16 * MT);

  // prefetch next M-group's A rows (lowers to global_prefetch_b8)
  __builtin_prefetch(A + (size_t)(m0 + 16 * MT + (lane & 15)) * K, 0, 1);

  v16h bfrag[KT];
  #pragma unroll
  for (int kt = 0; kt < KT; ++kt)
    bfrag[kt] = load_bT_frag(BT, K, n0, kt * 32, lane);

  int c = lane & 15, hi = lane >> 4;
  #pragma unroll
  for (int m = 0; m < MT; ++m) {
    int mm = m0 + m * 16;
    v8f acc = {};
    #pragma unroll
    for (int kt = 0; kt < KT; ++kt) {
      v16h a = load_a_frag(A, K, mm, kt * 32, lane);
      acc = __builtin_amdgcn_wmma_f32_16x16x32_f16(
          false, a, false, bfrag[kt], (short)0, acc, false, false);
    }
    #pragma unroll
    for (int i = 0; i < 8; ++i)
      C[(size_t)(mm + hi * 8 + i) * N + n0 + c] = acc[i];
  }
}

// out(Mx8,f32) = A(Mx128,f16) @ Wout + bout, Wout padded+transposed (16x128).
__global__ void gemm_out_kernel(const _Float16* __restrict__ A,
                                const _Float16* __restrict__ WpT,
                                const float* __restrict__ bout,
                                float* __restrict__ out) {
  int wave = threadIdx.x >> 5, lane = threadIdx.x & 31;
  int m0 = (blockIdx.x * 8 + wave) * 64;   // 4 M-tiles per wave

  v16h bfrag[4];
  #pragma unroll
  for (int kt = 0; kt < 4; ++kt)
    bfrag[kt] = load_bT_frag(WpT, DH, 0, kt * 32, lane);

  int c = lane & 15, hi = lane >> 4;
  float bias = (c < DOUT) ? bout[c] : 0.0f;

  #pragma unroll
  for (int m = 0; m < 4; ++m) {
    int mm = m0 + m * 16;
    v8f acc = {};
    #pragma unroll
    for (int kt = 0; kt < 4; ++kt) {
      v16h a = load_a_frag(A, DH, mm, kt * 32, lane);
      acc = __builtin_amdgcn_wmma_f32_16x16x32_f16(
          false, a, false, bfrag[kt], (short)0, acc, false, false);
    }
    if (c < DOUT) {
      #pragma unroll
      for (int i = 0; i < 8; ++i)
        out[(size_t)(mm + hi * 8 + i) * DOUT + c] = acc[i] + bias;
    }
  }
}

// ------------------------------- GCN aggregation ---------------------------
__global__ void aggregate_kernel(const float* __restrict__ h,
                                 const int* __restrict__ nbr,
                                 const float* __restrict__ dinv,
                                 float* __restrict__ agg) {
  int wave = threadIdx.x >> 5, lane = threadIdx.x & 31;
  int i = blockIdx.x * 8 + wave;
  float di = dinv[i];

  float hv[4];
  #pragma unroll
  for (int t = 0; t < 4; ++t) hv[t] = h[(size_t)i * DH + t * 32 + lane];

  for (int k = 0; k < KNN; ++k) {
    int j = nbr[(size_t)i * KNN + k];
    float coef = di * dinv[j];
    float* aj = agg + (size_t)j * DH + lane;
    #pragma unroll
    for (int t = 0; t < 4; ++t)
      __hip_atomic_fetch_add(aj + t * 32, hv[t] * coef, __ATOMIC_RELAXED,
                             __HIP_MEMORY_SCOPE_AGENT);
  }
}

// x_next = f16(tanh(agg + h*dinv^2 + bias))
__global__ void finalize_kernel(const float* __restrict__ h,
                                const float* __restrict__ agg,
                                const float* __restrict__ dinv,
                                const float* __restrict__ bias,
                                _Float16* __restrict__ xout) {
  size_t e = (size_t)blockIdx.x * 256 + threadIdx.x;
  int j = (int)(e >> 7);
  int d = (int)(e & (DH - 1));
  float di = dinv[j];
  float v = agg[e] + h[e] * di * di + bias[d];
  xout[e] = (_Float16)tanhf(v);
}

// ------------------------------- launcher ----------------------------------

extern "C" void kernel_launch(void* const* d_in, const int* in_sizes, int n_in,
                              void* d_out, int out_size, void* d_ws, size_t ws_size,
                              hipStream_t stream) {
  (void)in_sizes; (void)n_in; (void)out_size; (void)ws_size;

  const float* obs  = (const float*)d_in[0];   // (B,N,32)
  const float* W1   = (const float*)d_in[1];   // (32,128)
  const float* b1   = (const float*)d_in[2];   // (128,)
  const float* W2   = (const float*)d_in[3];   // (128,128)
  const float* b2   = (const float*)d_in[4];   // (128,)
  const float* Wout = (const float*)d_in[5];   // (128,8)
  const float* bout = (const float*)d_in[6];   // (8,)
  float* out = (float*)d_out;                  // (B,N,8)

  // workspace carve (all offsets multiples of 256B)
  char* w = (char*)d_ws;
  int*      nbr  = (int*)w;      w += (size_t)NNODES * KNN * 4;   // 8 MB
  float*    deg  = (float*)w;    w += (size_t)NNODES * 4;         // 0.5 MB
  float*    dinv = (float*)w;    w += (size_t)NNODES * 4;         // 0.5 MB
  _Float16* xh   = (_Float16*)w; w += (size_t)NNODES * DIN * 2;   // 8 MB
  _Float16* w1t  = (_Float16*)w; w += (size_t)DIN * DH * 2;       // 8 KB  (128x32)
  _Float16* w2t  = (_Float16*)w; w += (size_t)DH * DH * 2;        // 32 KB (128x128)
  _Float16* wpt  = (_Float16*)w; w += (size_t)16 * DH * 2;        // 4 KB  (16x128)
  float*    hbuf = (float*)w;    w += (size_t)NNODES * DH * 4;    // 64 MB
  float*    agg  = (float*)w;    w += (size_t)NNODES * DH * 4;    // 64 MB
  _Float16* xact = (_Float16*)w; w += (size_t)NNODES * DH * 2;    // 32 MB

  const int nElemH = NNODES * DH;       // 16777216
  const int nElemX = NNODES * DIN;      // 4194304

  // --- graph build ---
  zero_f32_kernel<<<NNODES / 256, 256, 0, stream>>>(deg, NNODES);
  knn_kernel<<<NB * (NN / 8), 256, 0, stream>>>(obs, nbr, deg);
  dinv_kernel<<<NNODES / 256, 256, 0, stream>>>(deg, dinv, NNODES);

  // --- f16 casts (weights transposed to N x K) ---
  cast_f32_f16_kernel<<<nElemX / 256, 256, 0, stream>>>(obs, xh, nElemX);
  transpose_cast_kernel<<<(DIN * DH) / 256, 256, 0, stream>>>(W1, w1t, DIN, DH);
  transpose_cast_kernel<<<(DH * DH) / 256, 256, 0, stream>>>(W2, w2t, DH, DH);
  pad_wout_t_kernel<<<(16 * DH) / 256, 256, 0, stream>>>(Wout, wpt);

  // --- layer 1: h = x @ W1 ; aggregate ; tanh ---
  gemm_h_kernel<1, 4><<<(NNODES / 64) * (DH / 16) / 8, 256, 0, stream>>>(
      xh, w1t, hbuf, DH, 3);
  zero_f32_kernel<<<nElemH / 256, 256, 0, stream>>>(agg, nElemH);
  aggregate_kernel<<<NNODES / 8, 256, 0, stream>>>(hbuf, nbr, dinv, agg);
  finalize_kernel<<<nElemH / 256, 256, 0, stream>>>(hbuf, agg, dinv, b1, xact);

  // --- layer 2: h = x2 @ W2 ; aggregate ; tanh ---
  gemm_h_kernel<4, 4><<<(NNODES / 64) * (DH / 16) / 8, 256, 0, stream>>>(
      xact, w2t, hbuf, DH, 3);
  zero_f32_kernel<<<nElemH / 256, 256, 0, stream>>>(agg, nElemH);
  aggregate_kernel<<<NNODES / 8, 256, 0, stream>>>(hbuf, nbr, dinv, agg);
  finalize_kernel<<<nElemH / 256, 256, 0, stream>>>(hbuf, agg, dinv, b2, xact);

  // --- output layer: out = x3 @ Wout + bout ---
  gemm_out_kernel<<<(NNODES / 64) / 8, 256, 0, stream>>>(xact, wpt, bout, out);
}
